// MoEGate_73753178407159
// MI455X (gfx1250) — matrix operand order repaired and checked by reading, not compiled, not run on hardware
//
#include <hip/hip_runtime.h>

typedef __attribute__((ext_vector_type(16))) __bf16          v16bf;
typedef __attribute__((ext_vector_type(16))) float           v16f;
typedef __attribute__((ext_vector_type(16))) unsigned short  v16us;
typedef __attribute__((ext_vector_type(8)))  float           v8f;

#define HDIM   2048
#define NEXP   64
#define TOPK   8
#define NBATCH 4
#define SEQLEN 4096
#define NTOK   (NBATCH * SEQLEN)        // 16384
#define KTILES (HDIM / 32)              // 64
#define TPB    64                       // tokens per block
#define WAVES  4
#define GATE_BLOCKS (NTOK / TPB)        // 256
#define BLOCKS_PER_BATCH (SEQLEN / TPB) // 64

// workspace layout (floats):
#define WS_SCORE 0     // [4][64] sum of softmax scores per (batch, expert)
#define WS_CE    256   // [4][64] top-k assignment counts per (batch, expert)
#define WS_Z     512   // scalar: sum of lse^2 over tokens
#define WS_NZERO 513
#define WS_WB_BYTES 4096                  // byte offset of bf16 weight fragments
#define WB_ELEMS (KTILES * 4 * 32 * 16)   // 131072 bf16 elements

#define WAIT_ASYNC(n) asm volatile("s_wait_asynccnt " #n ::: "memory")

__device__ __forceinline__ unsigned short f2bf_bits(float f) {
  unsigned u = __builtin_bit_cast(unsigned, f);
  u += 0x7FFFu + ((u >> 16) & 1u);          // round-to-nearest-even
  return (unsigned short)(u >> 16);
}

// Hardware B-slot (j, lane-half) -> logical K within a 32-wide tile, under the
// K-permutation sigma that makes A fragments contiguous per lane:
//   A lane-half h holds logical k = h*16 + 0..15 contiguously;
//   B slot j of half h must then hold the same contiguous k = h*16 + j.
__device__ __forceinline__ int b_slot_k(int j, int lane) {
  return j + ((lane >> 4) << 4);
}

// ---------------------------------------------------------------------------
// Kernel 1: gate weight [64 experts, 2048] fp32 -> bf16 in B-fragment order:
//   wb[((kt*4 + et)*32 + lane)*16 + j]
// so each lane's 16-element (32-byte) B fragment is contiguous.
// ---------------------------------------------------------------------------
__global__ void convert_weight_kernel(const float* __restrict__ w,
                                      unsigned short* __restrict__ wb) {
  int tid = blockIdx.x * blockDim.x + threadIdx.x;   // 0 .. WB_ELEMS-1
  int j    = tid & 15;
  int lane = (tid >> 4) & 31;
  int et   = (tid >> 9) & 3;
  int kt   = tid >> 11;
  int n    = et * 16 + (lane & 15);
  int k    = kt * 32 + b_slot_k(j, lane);
  wb[tid] = f2bf_bits(w[n * HDIM + k]);
}

__global__ void zero_ws_kernel(float* __restrict__ ws) {
  int tid = blockIdx.x * blockDim.x + threadIdx.x;
  if (tid < WS_NZERO) ws[tid] = 0.0f;
}

// ---------------------------------------------------------------------------
// Kernel 2: logits GEMM (bf16 WMMA, f32 acc) + softmax + top-8 + loss partials.
// 4 waves / block; each wave: 16 tokens x 64 experts, K-loop over 2048.
// A tiles are prefetched HBM -> LDS with async b128 copies, depth-2 pipeline
// over 4 LDS buffers (tracked with ASYNCcnt).
// ---------------------------------------------------------------------------
__global__ void __launch_bounds__(32 * WAVES)
gate_kernel(const float* __restrict__ x,
            const unsigned short* __restrict__ wb,
            int* __restrict__ idx_out,
            float* __restrict__ wt_out,
            float* __restrict__ scoreSum,   // [4][64]
            float* __restrict__ ceSum,      // [4][64]
            float* __restrict__ zSum) {
  __shared__ float ldsLog[TPB][NEXP + 1];   // +1 pad: conflict-free column reads
  __shared__ float tokMax[TPB];
  __shared__ float tokRcpSum[TPB];
  __shared__ float tokLse2[TPB];
  __shared__ float cnt[NEXP];
  // A staging: [wave][buf][lane][16 floats + pad]; 20-float (80B) lane stride
  // keeps the four 16B async chunks aligned, only 2-way LDS bank conflicts.
  __shared__ __align__(16) float abuf[WAVES][4][32][20];

  const int tid  = threadIdx.x;
  const int lane = tid & 31;
  const int w    = tid >> 5;
  const int rowBase = blockIdx.x * TPB + w * 16;
  const int m = rowBase + (lane & 15);      // token row this lane feeds into A
  const int kh = (lane >> 4) * 16;          // contiguous K half for this lane

  v8f acc0 = {}, acc1 = {}, acc2 = {}, acc3 = {};
  const v16us* bwv = (const v16us*)wb;
  const float* grow = x + (size_t)m * HDIM + kh;   // lane's base row pointer

  // issue one A tile (2KB/wave): 4 async b128 copies per lane, 64B contiguous
  auto issue_tile = [&](int kt, int buf) {
    const float* g = grow + kt * 32;
    unsigned l0 = (unsigned)(uintptr_t)&abuf[w][buf][lane][0];
    #pragma unroll
    for (int i = 0; i < 4; ++i) {
      asm volatile("global_load_async_to_lds_b128 %0, %1, off"
                   :: "v"(l0 + (unsigned)(i * 16)), "v"(g + i * 4)
                   : "memory");
    }
  };

  issue_tile(0, 0);
  issue_tile(1, 1);
  for (int kt = 0; kt < KTILES; ++kt) {
    if (kt + 2 < KTILES) {
      issue_tile(kt + 2, (kt + 2) & 3);
      WAIT_ASYNC(0x8);       // tile kt retired; kt+1, kt+2 still in flight
    } else if (kt + 1 < KTILES) {
      WAIT_ASYNC(0x4);       // tile kt retired; kt+1 in flight
    } else {
      WAIT_ASYNC(0x0);       // drain
    }

    // ---- A fragment: lane's 16 contiguous floats -> bf16 slots (RNE) ----
    const float* src = abuf[w][kt & 3][lane];
    v16f af;
    #pragma unroll
    for (int j = 0; j < 16; ++j) af[j] = src[j];
    v16bf a = __builtin_convertvector(af, v16bf);

    // ---- B fragments: 4 expert tiles, pre-packed contiguous 32B/lane ----
    v16bf b0 = __builtin_bit_cast(v16bf, bwv[(kt * 4 + 0) * 32 + lane]);
    v16bf b1 = __builtin_bit_cast(v16bf, bwv[(kt * 4 + 1) * 32 + lane]);
    v16bf b2 = __builtin_bit_cast(v16bf, bwv[(kt * 4 + 2) * 32 + lane]);
    v16bf b3 = __builtin_bit_cast(v16bf, bwv[(kt * 4 + 3) * 32 + lane]);

    acc0 = __builtin_amdgcn_wmma_f32_16x16x32_bf16(false, a, false, b0, (short)0, acc0, false, false);
    acc1 = __builtin_amdgcn_wmma_f32_16x16x32_bf16(false, a, false, b1, (short)0, acc1, false, false);
    acc2 = __builtin_amdgcn_wmma_f32_16x16x32_bf16(false, a, false, b2, (short)0, acc2, false, false);
    acc3 = __builtin_amdgcn_wmma_f32_16x16x32_bf16(false, a, false, b3, (short)0, acc3, false, false);
  }

  // ---- spill logits to LDS: C layout = lanes 0-15: M=r, lanes 16-31: M=r+8 ----
  {
    int tokHalf = (lane < 16) ? 0 : 8;
    int nc = lane & 15;
    #pragma unroll
    for (int r = 0; r < 8; ++r) {
      int tok = w * 16 + tokHalf + r;
      ldsLog[tok][0 * 16 + nc] = acc0[r];
      ldsLog[tok][1 * 16 + nc] = acc1[r];
      ldsLog[tok][2 * 16 + nc] = acc2[r];
      ldsLog[tok][3 * 16 + nc] = acc3[r];
    }
  }
  if (tid < NEXP) cnt[tid] = 0.0f;
  __syncthreads();

  // ---- phase 1: one thread per token — softmax stats, top-8, outputs ----
  if (tid < TPB) {
    float mx = -3.4e38f;
    #pragma unroll 8
    for (int e = 0; e < NEXP; ++e) mx = fmaxf(mx, ldsLog[tid][e]);
    float s = 0.0f;
    #pragma unroll 8
    for (int e = 0; e < NEXP; ++e) s += __expf(ldsLog[tid][e] - mx);
    float rcpS = 1.0f / s;
    float lse  = mx + __logf(s);
    tokMax[tid]    = mx;
    tokRcpSum[tid] = rcpS;
    tokLse2[tid]   = lse * lse;

    // stable descending top-8 on logits (monotone w/ softmax scores)
    float bv[TOPK];
    int   bi[TOPK];
    #pragma unroll
    for (int k = 0; k < TOPK; ++k) { bv[k] = -3.4e38f; bi[k] = 0; }
    for (int e = 0; e < NEXP; ++e) {
      float vv = ldsLog[tid][e];
      if (vv > bv[TOPK - 1]) {
        int p = TOPK - 1;
        while (p > 0 && vv > bv[p - 1]) { bv[p] = bv[p - 1]; bi[p] = bi[p - 1]; --p; }
        bv[p] = vv; bi[p] = e;
      }
    }
    float wv[TOPK];
    float denom = 0.0f;
    #pragma unroll
    for (int k = 0; k < TOPK; ++k) { wv[k] = __expf(bv[k] - mx) * rcpS; denom += wv[k]; }
    float rden = 1.0f / (denom + 1e-20f);   // NORM_TOPK_PROB, ROUTED_SCALE=1

    int tg = blockIdx.x * TPB + tid;
    #pragma unroll
    for (int k = 0; k < TOPK; ++k) {
      idx_out[tg * TOPK + k] = bi[k];
      wt_out[tg * TOPK + k]  = wv[k] * rden;
      atomicAdd(&cnt[bi[k]], 1.0f);
    }
  }
  __syncthreads();

  // ---- phase 2: one thread per expert — block-level reductions -> global ----
  int b = blockIdx.x / BLOCKS_PER_BATCH;
  if (tid < NEXP) {
    float s = 0.0f;
    for (int t = 0; t < TPB; ++t)
      s += __expf(ldsLog[t][tid] - tokMax[t]) * tokRcpSum[t];
    atomicAdd(&scoreSum[b * NEXP + tid], s);
    atomicAdd(&ceSum[b * NEXP + tid], cnt[tid]);
  }
  if (tid == 0) {
    float z = 0.0f;
    for (int t = 0; t < TPB; ++t) z += tokLse2[t];
    atomicAdd(zSum, z);
  }
}

// ---------------------------------------------------------------------------
// Kernel 3: fold partials into aux_loss scalar.
// ---------------------------------------------------------------------------
__global__ void finalize_kernel(const float* __restrict__ scoreSum,
                                const float* __restrict__ ceSum,
                                const float* __restrict__ zSum,
                                float* __restrict__ aux_out) {
  __shared__ float red[NEXP];
  int e = threadIdx.x;   // 64 threads
  float s = 0.0f;
  for (int b = 0; b < NBATCH; ++b) {
    float ce = ceSum[b * NEXP + e] * ((float)NEXP / ((float)SEQLEN * TOPK));
    float ms = scoreSum[b * NEXP + e] * (1.0f / (float)SEQLEN);
    s += ce * ms;
  }
  red[e] = s;
  __syncthreads();
  if (e == 0) {
    float t = 0.0f;
    for (int i = 0; i < NEXP; ++i) t += red[i];
    float z_loss = (zSum[0] / (float)NTOK) * 0.001f;
    aux_out[0] = t / (float)NBATCH + z_loss;
  }
}

extern "C" void kernel_launch(void* const* d_in, const int* in_sizes, int n_in,
                              void* d_out, int out_size, void* d_ws, size_t ws_size,
                              hipStream_t stream) {
  const float* x   = (const float*)d_in[0];   // [4, 4096, 2048]
  const float* wgt = (const float*)d_in[1];   // [64, 2048]

  float* wsF = (float*)d_ws;
  unsigned short* wb = (unsigned short*)((char*)d_ws + WS_WB_BYTES);

  int*   idx_out = (int*)d_out;                          // [16384, 8] int32
  float* wt_out  = (float*)d_out + NTOK * TOPK;          // [16384, 8] f32
  float* aux_out = (float*)d_out + 2 * NTOK * TOPK;      // scalar

  convert_weight_kernel<<<WB_ELEMS / 256, 256, 0, stream>>>(wgt, wb);
  zero_ws_kernel<<<3, 256, 0, stream>>>(wsF);
  gate_kernel<<<GATE_BLOCKS, 32 * WAVES, 0, stream>>>(
      x, wb, idx_out, wt_out, wsF + WS_SCORE, wsF + WS_CE, wsF + WS_Z);
  finalize_kernel<<<1, NEXP, 0, stream>>>(wsF + WS_SCORE, wsF + WS_CE,
                                          wsF + WS_Z, aux_out);
}
